// MPN_89369679495448
// MI455X (gfx1250) — compile-verified, matrix operand order checked by reading, not run on hardware
//
#include <hip/hip_runtime.h>

typedef _Float16 half_t;
typedef __attribute__((ext_vector_type(16))) _Float16 v16h;
typedef __attribute__((ext_vector_type(8)))  float    v8f;

namespace {
constexpr int HIDDEN = 300;
constexpr int HP     = 320;          // padded hidden: 20x16 N-tiles, 10x32 K-chunks
constexpr int NHEADS = 4;
constexpr int NEDGE  = 60000;        // multiple of 32
constexpr int NATOM  = 30000;
constexpr int NATOMP = 30016;        // padded to multiple of 32 for 32-row GEMM tiles
constexpr int MAXNB  = 6;
constexpr int NMOL   = 600;
constexpr int APM    = 50;
constexpr int AFD    = 144;
constexpr int BONDIN = 158;
constexpr int BINP   = 160;          // bond-input padded
constexpr int AINP   = 480;          // [fatoms 0..143 | pad | a_nei 160..459 | pad]
constexpr int COMPLD = NHEADS * HP;  // 1280
}

// ---------- conversion / padding kernels ----------
__global__ void k_cvt_pad(const float* __restrict__ src, half_t* __restrict__ dst,
                          int R, int C, int Rp, int Cp) {
  int idx = blockIdx.x * blockDim.x + threadIdx.x;
  int total = Rp * Cp;
  if (idx >= total) return;
  int r = idx / Cp, c = idx - r * Cp;
  float v = (r < R && c < C) ? src[(size_t)r * C + c] : 0.f;
  dst[idx] = (half_t)v;
}

__global__ void k_cvt_wh(const float* __restrict__ src, half_t* __restrict__ dst) {
  // src [300][1200] with cols head-major (n*300+h) -> dst [320][1280] cols n*320+h
  int idx = blockIdx.x * blockDim.x + threadIdx.x;
  if (idx >= HP * COMPLD) return;
  int r = idx / COMPLD, c = idx - r * COMPLD;
  int n = c / HP, h = c - n * HP;
  float v = (r < HIDDEN && h < HIDDEN)
                ? src[(size_t)r * (NHEADS * HIDDEN) + n * HIDDEN + h] : 0.f;
  dst[idx] = (half_t)v;
}

__global__ void k_cvt_wo(const float* __restrict__ src, half_t* __restrict__ dst) {
  // src [300][444] (in = fatoms144 ++ a_nei300) -> dst [320][480]
  int idx = blockIdx.x * blockDim.x + threadIdx.x;
  if (idx >= HP * AINP) return;
  int r = idx / AINP, c = idx - r * AINP;
  float v = 0.f;
  if (r < HIDDEN) {
    if (c < AFD)                  v = src[(size_t)r * (AFD + HIDDEN) + c];
    else if (c >= 160 && c < 460) v = src[(size_t)r * (AFD + HIDDEN) + (c - 16)];
  }
  dst[idx] = (half_t)v;
}

// ---------- WMMA GEMM, 2x2 register tile: 32x32 C per wave ----------
// C[M x 320] = A[M x K] * B[320 x K]^T ; 4 wmma per 8 b128 loads (16 FLOP/B from L2)
union FragU { v16h v; uint4 q[2]; };

__global__ void __launch_bounds__(256)
k_gemm(const half_t* __restrict__ A, const half_t* __restrict__ B,
       half_t* __restrict__ Ch, half_t* __restrict__ Ch2, float* __restrict__ Cf,
       const half_t* __restrict__ auxh, const float* __restrict__ bias,
       int M, int K, int mode, size_t bStrideY, size_t cStrideY)
{
  const int NT2  = HP / 32;                          // 10 n-tile pairs
  const int wave = blockIdx.x * 8 + (threadIdx.x >> 5);
  const int total = (M >> 5) * NT2;
  if (wave >= total) return;                         // wave-uniform; EXEC stays all-1s

  B += (size_t)blockIdx.y * bStrideY;                // head batching for W_ma
  const size_t cOff = (size_t)blockIdx.y * cStrideY;

  const int mt2  = wave / NT2;
  const int nt2  = wave - mt2 * NT2;
  const int lane = threadIdx.x & 31;
  const int lrow = lane & 15;
  const int koff = (lane >> 4) << 3;                 // K halves 0 / 8 (A-layout)

  const half_t* pa0 = A + (size_t)(mt2 * 32 + lrow) * K + koff;
  const half_t* pa1 = pa0 + (size_t)16 * K;
  const half_t* pb0 = B + (size_t)(nt2 * 32 + lrow) * K + koff;
  const half_t* pb1 = pb0 + (size_t)16 * K;

  v8f acc[4] = {v8f{}, v8f{}, v8f{}, v8f{}};         // [mi*2+ni]
  for (int k0 = 0; k0 < K; k0 += 32) {
    FragU a0, a1, b0, b1;
    a0.q[0] = *(const uint4*)(pa0 + k0);
    a0.q[1] = *(const uint4*)(pa0 + k0 + 16);
    a1.q[0] = *(const uint4*)(pa1 + k0);
    a1.q[1] = *(const uint4*)(pa1 + k0 + 16);
    b0.q[0] = *(const uint4*)(pb0 + k0);
    b0.q[1] = *(const uint4*)(pb0 + k0 + 16);
    b1.q[0] = *(const uint4*)(pb1 + k0);
    b1.q[1] = *(const uint4*)(pb1 + k0 + 16);
    acc[0] = __builtin_amdgcn_wmma_f32_16x16x32_f16(false, a0.v, false, b0.v,
                                                    (short)0, acc[0], false, false);
    acc[1] = __builtin_amdgcn_wmma_f32_16x16x32_f16(false, a0.v, false, b1.v,
                                                    (short)0, acc[1], false, false);
    acc[2] = __builtin_amdgcn_wmma_f32_16x16x32_f16(false, a1.v, false, b0.v,
                                                    (short)0, acc[2], false, false);
    acc[3] = __builtin_amdgcn_wmma_f32_16x16x32_f16(false, a1.v, false, b1.v,
                                                    (short)0, acc[3], false, false);
  }

  const int ncb = nt2 * 32 + lrow;                   // C layout: lane&15 -> col
  const int mbb = mt2 * 32 + ((lane >> 4) << 3);     // lane>>4 -> +8 rows
  #pragma unroll
  for (int t = 0; t < 4; ++t) {
    const int mb = mbb + (t >> 1) * 16;
    const int nc = ncb + (t & 1) * 16;
    #pragma unroll
    for (int r = 0; r < 8; ++r) {
      const size_t off = cOff + (size_t)(mb + r) * HP + nc;
      float v = acc[t][r];
      if (mode == 0) {
        Ch[off] = (half_t)v;
      } else if (mode == 1) {                        // binput raw + message = relu
        Ch2[off] = (half_t)v;
        Ch[off]  = (half_t)fmaxf(v, 0.f);
      } else if (mode == 2) {                        // relu(acc + binput)
        v += (float)auxh[off];
        Ch[off] = (half_t)fmaxf(v, 0.f);
      } else if (mode == 3) {                        // relu(acc + bias) -> fp16
        v += (nc < HIDDEN) ? bias[nc] : 0.f;
        Ch[off] = (half_t)fmaxf(v, 0.f);
      } else {                                       // relu(acc + bias) -> fp32
        v += (nc < HIDDEN) ? bias[nc] : 0.f;
        Cf[off] = fmaxf(v, 0.f);
      }
    }
  }
}

// ---------- fused edge attention: scores + softmax + combine ----------
__global__ void __launch_bounds__(256)
k_attn(const half_t* __restrict__ msg, const half_t* __restrict__ mt,
       const int* __restrict__ bgraph, half_t* __restrict__ comp)
{
  __shared__ half_t sNei[8][MAXNB][HP];              // 30 KB
  const int w    = threadIdx.x >> 5;
  const int lane = threadIdx.x & 31;
  const int e    = blockIdx.x * 8 + w;
  const int c0   = lane * 10;                        // 320 / 32 lanes

  float mv[NHEADS][10];
  #pragma unroll
  for (int n = 0; n < NHEADS; ++n) {
    const half_t* pm = mt + ((size_t)n * NEDGE + e) * HP + c0;
    #pragma unroll
    for (int i = 0; i < 10; ++i) mv[n][i] = (float)pm[i];
  }

  int nb[MAXNB];
  #pragma unroll
  for (int k = 0; k < MAXNB; ++k) nb[k] = bgraph[e * MAXNB + k];

  float part[NHEADS][MAXNB];
  #pragma unroll
  for (int n = 0; n < NHEADS; ++n)
    #pragma unroll
    for (int k = 0; k < MAXNB; ++k) part[n][k] = 0.f;

  #pragma unroll
  for (int k = 0; k < MAXNB; ++k) {
    const half_t* pn = msg + (size_t)nb[k] * HP + c0;
    #pragma unroll
    for (int i = 0; i < 10; ++i) {
      half_t hv = pn[i];
      sNei[w][k][c0 + i] = hv;                       // same-lane RAW only
      float x = (float)hv;
      #pragma unroll
      for (int n = 0; n < NHEADS; ++n) part[n][k] += x * mv[n][i];
    }
  }

  #pragma unroll
  for (int n = 0; n < NHEADS; ++n)
    #pragma unroll
    for (int k = 0; k < MAXNB; ++k) {
      float v = part[n][k];
      #pragma unroll
      for (int off = 16; off > 0; off >>= 1) v += __shfl_xor(v, off, 32);
      part[n][k] = v;                                // replicated on all lanes
    }

  float wgt[NHEADS][MAXNB];
  #pragma unroll
  for (int n = 0; n < NHEADS; ++n) {
    float mx = part[n][0];
    #pragma unroll
    for (int k = 1; k < MAXNB; ++k) mx = fmaxf(mx, part[n][k]);
    float s = 0.f;
    #pragma unroll
    for (int k = 0; k < MAXNB; ++k) { float t = __expf(part[n][k] - mx); wgt[n][k] = t; s += t; }
    float inv = 1.f / s;
    #pragma unroll
    for (int k = 0; k < MAXNB; ++k) wgt[n][k] *= inv;
  }

  half_t* pc = comp + (size_t)e * COMPLD + c0;
  #pragma unroll
  for (int i = 0; i < 10; ++i) {
    #pragma unroll
    for (int n = 0; n < NHEADS; ++n) {
      float a = 0.f;
      #pragma unroll
      for (int k = 0; k < MAXNB; ++k) a += wgt[n][k] * (float)sNei[w][k][c0 + i];
      pc[(size_t)n * HP + i] = (half_t)a;
    }
  }
}

// ---------- atom gather-sum + padded ainput build ----------
__global__ void __launch_bounds__(256)
k_anei(const float* __restrict__ fatoms, const half_t* __restrict__ msg,
       const int* __restrict__ agraph, half_t* __restrict__ ainput)
{
  const int a    = blockIdx.x * 8 + (threadIdx.x >> 5);
  const int lane = threadIdx.x & 31;
  if (a >= NATOM) return;
  int nb[MAXNB];
  #pragma unroll
  for (int k = 0; k < MAXNB; ++k) nb[k] = agraph[a * MAXNB + k];
  half_t* dst = ainput + (size_t)a * AINP;
  for (int c = lane; c < AINP; c += 32) {
    float v = 0.f;
    if (c < AFD) v = fatoms[(size_t)a * AFD + c];
    else if (c >= 160 && c < 460) {
      int h = c - 160;
      #pragma unroll
      for (int k = 0; k < MAXNB; ++k) v += (float)msg[(size_t)nb[k] * HP + h];
    }
    dst[c] = (half_t)v;
  }
}

// ---------- per-molecule attention (wave per (mol, atom)) ----------
__global__ void __launch_bounds__(256)
k_molattn(const half_t* __restrict__ curA, const half_t* __restrict__ cur,
          half_t* __restrict__ attc)
{
  const int w    = blockIdx.x * 8 + (threadIdx.x >> 5);
  const int lane = threadIdx.x & 31;
  if (w >= NMOL * APM) return;
  const int m  = w / APM;
  const int c0 = lane * 10;

  float qa[10];
  const half_t* pq = curA + (size_t)w * HP + c0;
  #pragma unroll
  for (int i = 0; i < 10; ++i) qa[i] = (float)pq[i];

  const half_t* base = cur + (size_t)m * APM * HP;
  float s[APM];
  #pragma unroll 2
  for (int b = 0; b < APM; ++b) {
    const half_t* pb = base + (size_t)b * HP + c0;
    float p = 0.f;
    #pragma unroll
    for (int i = 0; i < 10; ++i) p += qa[i] * (float)pb[i];
    #pragma unroll
    for (int off = 16; off > 0; off >>= 1) p += __shfl_xor(p, off, 32);
    s[b] = p;
  }
  float mx = s[0];
  for (int b = 1; b < APM; ++b) mx = fmaxf(mx, s[b]);
  float sum = 0.f;
  for (int b = 0; b < APM; ++b) { s[b] = __expf(s[b] - mx); sum += s[b]; }
  float inv = 1.f / sum;

  half_t* pout = attc + (size_t)w * HP + c0;
  #pragma unroll
  for (int i = 0; i < 10; ++i) {
    float a = 0.f;
    #pragma unroll 2
    for (int b = 0; b < APM; ++b) a += s[b] * (float)base[(size_t)b * HP + c0 + i];
    pout[i] = (half_t)(a * inv);
  }
}

// ---------- final per-molecule mean ----------
__global__ void k_final(const half_t* __restrict__ cur, const float* __restrict__ attb,
                        float* __restrict__ out)
{
  const int m = blockIdx.x;
  const int h = threadIdx.x;
  if (h >= HIDDEN) return;
  float s = 0.f;
  for (int a = 0; a < APM; ++a) {
    size_t off = ((size_t)(m * APM + a)) * HP + h;
    s += (float)cur[off] + attb[off];
  }
  out[m * HIDDEN + h] = s * (1.0f / APM);
}

extern "C" void kernel_launch(void* const* d_in, const int* in_sizes, int n_in,
                              void* d_out, int out_size, void* d_ws, size_t ws_size,
                              hipStream_t stream) {
  const float* fatoms = (const float*)d_in[0];
  const float* fbonds = (const float*)d_in[1];
  const int*   agraph = (const int*)d_in[2];
  const int*   bgraph = (const int*)d_in[3];
  const float* W_i    = (const float*)d_in[4];
  const float* W_ma   = (const float*)d_in[5];
  const float* W_h    = (const float*)d_in[6];
  const float* W_o    = (const float*)d_in[7];
  const float* b_o    = (const float*)d_in[8];
  const float* W_a    = (const float*)d_in[9];
  const float* W_b    = (const float*)d_in[10];
  const float* b_b    = (const float*)d_in[11];
  float* out = (float*)d_out;
  (void)in_sizes; (void)n_in; (void)out_size; (void)ws_size;

  // workspace carve-out (~530 MB total)
  char* p = (char*)d_ws;
  auto alloc = [&](size_t bytes) -> void* {
    void* r = (void*)p; p += (bytes + 255) & ~(size_t)255; return r;
  };
  half_t* fbonds_h = (half_t*)alloc((size_t)NEDGE * BINP * 2);
  half_t* Wi_h     = (half_t*)alloc((size_t)HP * BINP * 2);
  half_t* Wma_h    = (half_t*)alloc((size_t)NHEADS * HP * HP * 2);
  half_t* Wh_h     = (half_t*)alloc((size_t)HP * COMPLD * 2);
  half_t* Wo_h     = (half_t*)alloc((size_t)HP * AINP * 2);
  half_t* Wa_h     = (half_t*)alloc((size_t)HP * HP * 2);
  half_t* Wb_h     = (half_t*)alloc((size_t)HP * HP * 2);
  half_t* binput_h = (half_t*)alloc((size_t)NEDGE * HP * 2);
  half_t* msg_h    = (half_t*)alloc((size_t)NEDGE * HP * 2);
  half_t* mt_h     = (half_t*)alloc((size_t)NHEADS * NEDGE * HP * 2);
  half_t* comp_h   = (half_t*)alloc((size_t)NEDGE * COMPLD * 2);
  half_t* ain_h    = (half_t*)alloc((size_t)NATOMP * AINP * 2);   // padded rows
  half_t* cur_h    = (half_t*)alloc((size_t)NATOMP * HP * 2);
  half_t* curA_h   = (half_t*)alloc((size_t)NATOMP * HP * 2);
  half_t* attc_h   = (half_t*)alloc((size_t)NATOMP * HP * 2);
  float*  attb_f   = (float*)alloc((size_t)NATOMP * HP * 4);

  auto cvt = [&](const float* s, half_t* d, int R, int C, int Rp, int Cp) {
    int total = Rp * Cp;
    k_cvt_pad<<<(total + 255) / 256, 256, 0, stream>>>(s, d, R, C, Rp, Cp);
  };
  cvt(fbonds, fbonds_h, NEDGE, BONDIN, NEDGE, BINP);
  cvt(W_i, Wi_h, HIDDEN, BONDIN, HP, BINP);
  for (int n = 0; n < NHEADS; ++n)
    cvt(W_ma + (size_t)n * HIDDEN * HIDDEN, Wma_h + (size_t)n * HP * HP,
        HIDDEN, HIDDEN, HP, HP);
  k_cvt_wh<<<(HP * COMPLD + 255) / 256, 256, 0, stream>>>(W_h, Wh_h);
  k_cvt_wo<<<(HP * AINP + 255) / 256, 256, 0, stream>>>(W_o, Wo_h);
  cvt(W_a, Wa_h, HIDDEN, HIDDEN, HP, HP);
  cvt(W_b, Wb_h, HIDDEN, HIDDEN, HP, HP);

  auto gemmGrid = [](int M, int heads) {
    int tiles = (M / 32) * (HP / 32);                // 32x32 tiles per wave
    return dim3((unsigned)((tiles + 7) / 8), (unsigned)heads, 1);
  };

  // binput = fbonds @ W_i^T ; message = relu(binput)
  k_gemm<<<gemmGrid(NEDGE, 1), 256, 0, stream>>>(
      fbonds_h, Wi_h, msg_h, binput_h, nullptr, nullptr, nullptr,
      NEDGE, BINP, 1, 0, 0);

  for (int it = 0; it < 3; ++it) {                   // DEPTH-1 iterations
    // mt[n] = message @ W_ma[n]^T  (head-batched over gridDim.y)
    k_gemm<<<gemmGrid(NEDGE, NHEADS), 256, 0, stream>>>(
        msg_h, Wma_h, mt_h, nullptr, nullptr, nullptr, nullptr,
        NEDGE, HP, 0, (size_t)HP * HP, (size_t)NEDGE * HP);
    // scores -> softmax -> comp
    k_attn<<<NEDGE / 8, 256, 0, stream>>>(msg_h, mt_h, bgraph, comp_h);
    // message = relu(binput + comp @ W_h^T)
    k_gemm<<<gemmGrid(NEDGE, 1), 256, 0, stream>>>(
        comp_h, Wh_h, msg_h, nullptr, nullptr, binput_h, nullptr,
        NEDGE, COMPLD, 2, 0, 0);
  }

  // ainput = [fatoms | gather-sum(message)] ; cur = relu(ainput @ W_o^T + b_o)
  k_anei<<<(NATOM + 7) / 8, 256, 0, stream>>>(fatoms, msg_h, agraph, ain_h);
  k_gemm<<<gemmGrid(NATOMP, 1), 256, 0, stream>>>(
      ain_h, Wo_h, cur_h, nullptr, nullptr, nullptr, b_o, NATOMP, AINP, 3, 0, 0);

  // molecule attention
  k_gemm<<<gemmGrid(NATOMP, 1), 256, 0, stream>>>(
      cur_h, Wa_h, curA_h, nullptr, nullptr, nullptr, nullptr, NATOMP, HP, 0, 0, 0);
  k_molattn<<<(NMOL * APM) / 8, 256, 0, stream>>>(curA_h, cur_h, attc_h);
  k_gemm<<<gemmGrid(NATOMP, 1), 256, 0, stream>>>(
      attc_h, Wb_h, nullptr, nullptr, attb_f, nullptr, b_b, NATOMP, HP, 4, 0, 0);

  // mol_vecs = mean_a(cur + att_h)
  k_final<<<NMOL, HP, 0, stream>>>(cur_h, attb_f, out);
}
